// GatedFeedbackLSTM_66941360275695
// MI455X (gfx1250) — matrix operand order repaired and checked by reading, not compiled, not run on hardware
//
#include <hip/hip_runtime.h>
#include <hip/hip_bf16.h>

// ---------------- problem constants ----------------
#define B_   64
#define T_   256
#define D_   512
#define H_   512
#define NL_  3
#define BH   (B_ * H_)          // 32768
#define NWG  64
#define NTHR 256

// workspace element counts (bf16 halves)
constexpr size_t N_BX   = (size_t)B_ * T_ * D_;        // 8388608
constexpr size_t N_WI2H = (size_t)NL_ * 3 * H_ * D_;   // 2359296
constexpr size_t N_WH2H = N_WI2H;
constexpr size_t N_WWC  = (size_t)NL_ * H_ * D_;       // 786432
constexpr size_t N_WWG  = (size_t)NL_ * NL_ * H_ * D_; // 2359296
constexpr size_t N_WUG  = (size_t)NL_ * NL_ * H_ * (H_ * NL_); // 7077888
constexpr size_t N_WUC  = (size_t)NL_ * NL_ * H_ * H_; // 2359296
constexpr size_t N_HBF  = (size_t)2 * NL_ * BH;        // 196608

typedef __attribute__((ext_vector_type(16))) __bf16 v16bf;
typedef __attribute__((ext_vector_type(8)))  float  v8f;

struct alignas(16) U4 { unsigned x, y, z, w; };
union Frag { U4 q[2]; v16bf v; };

__device__ __forceinline__ unsigned short f2bf(float f) {
    unsigned u = __float_as_uint(f);
    u += 0x7FFFu + ((u >> 16) & 1u);        // round-to-nearest-even
    return (unsigned short)(u >> 16);
}

__device__ __forceinline__ float sigmoidf_(float x) {
    return 1.0f / (1.0f + __expf(-x));
}

// Monotonic ticket grid barrier (counter zeroed by hipMemsetAsync each call).
__device__ __forceinline__ void grid_barrier(unsigned* bar) {
    __syncthreads();
    if (threadIdx.x == 0) {
        __threadfence();
        unsigned ticket = __hip_atomic_fetch_add(bar, 1u, __ATOMIC_RELAXED,
                                                 __HIP_MEMORY_SCOPE_AGENT) + 1u;
        unsigned target = ((ticket + NWG - 1u) / NWG) * NWG;
        while (__hip_atomic_load(bar, __ATOMIC_RELAXED,
                                 __HIP_MEMORY_SCOPE_AGENT) < target) {
            __builtin_amdgcn_s_sleep(1);
        }
        __threadfence();
    }
    __syncthreads();
}

__device__ __forceinline__ v8f wmma_bf16(const Frag& a, const Frag& b, v8f c) {
    return __builtin_amdgcn_wmma_f32_16x16x32_bf16(
        false, a.v, false, b.v, (short)0, c, false, false);
}

// A fragment per ISA 16-bit 16x32 layout (lane m=lane&15, K chunks at hiA / 16+hiA);
// B fragment per ISA 32x16 layout (lane n=lane&15, K chunks at hiB / 16+hiB).
__device__ __forceinline__ void frag_loadA(Frag& f, const unsigned short* a, int k) {
    f.q[0] = *(const U4*)(a + k);
    f.q[1] = *(const U4*)(a + k + 16);
}
__device__ __forceinline__ void frag_loadB(Frag& f, const unsigned short* b, int k) {
    f.q[0] = *(const U4*)(b + k);
    f.q[1] = *(const U4*)(b + k + 16);
}

// 32x32 output block: acc[mi][ni] += A[m0+16mi, :K] * B[n0+16ni, :K]^T
__device__ __forceinline__ void gemm2x2(v8f& c00, v8f& c01, v8f& c10, v8f& c11,
                                        const unsigned short* __restrict__ A, int lda,
                                        const unsigned short* __restrict__ B, int ldb,
                                        int K, int m0, int n0, int lane) {
    const int mn  = lane & 15;
    const int hiA = (lane >> 4) << 3;   // 0 or 8
    const int hiB = (lane >> 4) << 3;   // 0 or 8
    const unsigned short* a0 = A + (size_t)(m0 + mn) * lda + hiA;
    const unsigned short* a1 = a0 + (size_t)16 * lda;
    const unsigned short* b0 = B + (size_t)(n0 + mn) * ldb + hiB;
    const unsigned short* b1 = b0 + (size_t)16 * ldb;
#pragma unroll 2
    for (int k = 0; k < K; k += 32) {
        Frag fa0, fa1, fb0, fb1;
        frag_loadA(fa0, a0, k);
        frag_loadA(fa1, a1, k);
        frag_loadB(fb0, b0, k);
        frag_loadB(fb1, b1, k);
        c00 = wmma_bf16(fa0, fb0, c00);
        c01 = wmma_bf16(fa0, fb1, c01);
        c10 = wmma_bf16(fa1, fb0, c10);
        c11 = wmma_bf16(fa1, fb1, c11);
    }
}

// 16x32 output block: acc[ni] += A[m0, :K] * B[n0+16ni, :K]^T
__device__ __forceinline__ void gemm1x2(v8f& c0, v8f& c1,
                                        const unsigned short* __restrict__ A, int lda,
                                        const unsigned short* __restrict__ B, int ldb,
                                        int K, int m0, int n0, int lane) {
    const int mn  = lane & 15;
    const int hiA = (lane >> 4) << 3;
    const int hiB = (lane >> 4) << 3;
    const unsigned short* a0 = A + (size_t)(m0 + mn) * lda + hiA;
    const unsigned short* b0 = B + (size_t)(n0 + mn) * ldb + hiB;
    const unsigned short* b1 = b0 + (size_t)16 * ldb;
#pragma unroll 2
    for (int k = 0; k < K; k += 32) {
        Frag fa, fb0, fb1;
        frag_loadA(fa, a0, k);
        frag_loadB(fb0, b0, k);
        frag_loadB(fb1, b1, k);
        c0 = wmma_bf16(fa, fb0, c0);
        c1 = wmma_bf16(fa, fb1, c1);
    }
}

// D layout: VGPR v -> row m0 + v + 8*(lane>=16), col n0 + (lane&15)
__device__ __forceinline__ void store_tile(float* __restrict__ dst, int pitch,
                                           int m0, int n0, v8f acc, int lane) {
    const int hi8 = (lane >> 4) << 3;
    const int col = n0 + (lane & 15);
#pragma unroll
    for (int v = 0; v < 8; ++v)
        dst[(size_t)(m0 + hi8 + v) * pitch + col] = acc[v];
}

__global__ __launch_bounds__(NTHR) void GatedFeedbackLSTM_persistent(
    const float* __restrict__ x,    const float* __restrict__ h0,
    const float* __restrict__ c0,
    const float* __restrict__ Wi2h, const float* __restrict__ bi2h,
    const float* __restrict__ Wh2h, const float* __restrict__ bh2h,
    const float* __restrict__ Wwc,  const float* __restrict__ bwc,
    const float* __restrict__ Wwg,  const float* __restrict__ bwg,
    const float* __restrict__ Wug,  const float* __restrict__ bug,
    const float* __restrict__ Wuc,  const float* __restrict__ buc,
    float* __restrict__ out, char* __restrict__ ws)
{
    // ---- workspace carve-up ----
    unsigned*       bar  = (unsigned*)ws;
    unsigned short* bx   = (unsigned short*)(ws + 128);
    unsigned short* wi2h = bx   + N_BX;
    unsigned short* wh2h = wi2h + N_WI2H;
    unsigned short* wwc  = wh2h + N_WH2H;
    unsigned short* wwg  = wwc  + N_WWC;
    unsigned short* wug  = wwg  + N_WWG;
    unsigned short* wuc  = wug  + N_WUG;
    unsigned short* hbf  = wuc  + N_WUC;                 // 2 * NL * BH (ping/pong)
    float* h_state   = (float*)(hbf + N_HBF);            // NL*BH
    float* c_state   = h_state + (size_t)NL_ * BH;
    float* pre_gates = c_state + (size_t)NL_ * BH;       // 64 x 1536
    float* gg_pre    = pre_gates + (size_t)B_ * 3 * H_;  // 3 x BH
    float* uc_pre    = gg_pre + (size_t)NL_ * BH;        // 3 x BH
    float* wc_pre    = uc_pre + (size_t)NL_ * BH;        // BH

    const int tid      = threadIdx.x;
    const int gtid     = blockIdx.x * NTHR + tid;
    const int gthreads = NWG * NTHR;                     // 16384
    const int lane     = tid & 31;
    const int wave     = gtid >> 5;                      // 0..511
    const int nwaves   = gthreads >> 5;

    // ---- phase 0: fp32 -> bf16 conversion of x, weights; state init ----
    for (size_t i = gtid; i < N_BX;   i += gthreads) bx[i]   = f2bf(x[i]);
    for (size_t i = gtid; i < N_WI2H; i += gthreads) wi2h[i] = f2bf(Wi2h[i]);
    for (size_t i = gtid; i < N_WH2H; i += gthreads) wh2h[i] = f2bf(Wh2h[i]);
    for (size_t i = gtid; i < N_WWC;  i += gthreads) wwc[i]  = f2bf(Wwc[i]);
    for (size_t i = gtid; i < N_WWG;  i += gthreads) wwg[i]  = f2bf(Wwg[i]);
    for (size_t i = gtid; i < N_WUG;  i += gthreads) wug[i]  = f2bf(Wug[i]);
    for (size_t i = gtid; i < N_WUC;  i += gthreads) wuc[i]  = f2bf(Wuc[i]);
    for (size_t i = gtid; i < (size_t)NL_ * BH; i += gthreads) {
        hbf[i]     = f2bf(h0[i]);
        h_state[i] = h0[i];
        c_state[i] = c0[i];
    }
    grid_barrier(bar);

    // ---- recurrence ----
    for (int t = 0; t < T_; ++t) {
        const unsigned short* hprev = hbf + (size_t)(t & 1) * NL_ * BH;
        unsigned short*       hnext = hbf + (size_t)((t + 1) & 1) * NL_ * BH;

        for (int L = 0; L < NL_; ++L) {
            const unsigned short* inpA;
            int lda;
            if (L == 0) { inpA = bx + (size_t)t * D_; lda = T_ * D_; }
            else        { inpA = hnext + (size_t)(L - 1) * BH; lda = H_; }

            // ---- GEMM phase: 416 blocked jobs, <=1 per wave ----
            for (int job = wave; job < 416; job += nwaves) {
                if (job < 96) {
                    // gates [64 x 1536], 2x2 blocking: 96 jobs x 128 WMMA
                    const int mp = job & 1, np = job >> 1;     // np in 0..47
                    const int m0 = mp * 32, n0 = np * 32;
                    v8f c00 = {}, c01 = {}, c10 = {}, c11 = {};
                    gemm2x2(c00, c01, c10, c11, inpA, lda,
                            wi2h + (size_t)L * 1536 * 512, 512, 512, m0, n0, lane);
                    gemm2x2(c00, c01, c10, c11, hprev + (size_t)L * BH, H_,
                            wh2h + (size_t)L * 1536 * 512, 512, 512, m0, n0, lane);
                    store_tile(pre_gates, 1536, m0,      n0,      c00, lane);
                    store_tile(pre_gates, 1536, m0,      n0 + 16, c01, lane);
                    store_tile(pre_gates, 1536, m0 + 16, n0,      c10, lane);
                    store_tile(pre_gates, 1536, m0 + 16, n0 + 16, c11, lane);
                } else if (job < 288) {
                    // gg_pre [3][64 x 512], 1x2 blocking: 192 jobs x 128 WMMA
                    const int idx = job - 96;
                    const int mt = idx & 3, rest = idx >> 2;   // rest in 0..47
                    const int ls = rest >> 4, np = rest & 15;
                    const int m0 = mt * 16, n0 = np * 32;
                    v8f c0 = {}, c1 = {};
                    gemm1x2(c0, c1, inpA, lda,
                            wwg + (size_t)(L * 3 + ls) * 512 * 512, 512, 512, m0, n0, lane);
#pragma unroll
                    for (int lk = 0; lk < NL_; ++lk)           // K=1536 split 3x512
                        gemm1x2(c0, c1, hprev + (size_t)lk * BH, H_,
                                wug + (size_t)(L * 3 + ls) * 512 * 1536 + (size_t)lk * 512,
                                1536, 512, m0, n0, lane);
                    store_tile(gg_pre + (size_t)ls * BH, 512, m0, n0,      c0, lane);
                    store_tile(gg_pre + (size_t)ls * BH, 512, m0, n0 + 16, c1, lane);
                } else if (job < 384) {
                    // uc_pre [3][64 x 512], 2x2 blocking: 96 jobs x 64 WMMA
                    const int idx = job - 288;
                    const int mp = idx & 1, rest = idx >> 1;   // rest in 0..47
                    const int ls = rest >> 4, np = rest & 15;
                    const int m0 = mp * 32, n0 = np * 32;
                    v8f c00 = {}, c01 = {}, c10 = {}, c11 = {};
                    gemm2x2(c00, c01, c10, c11, hprev + (size_t)ls * BH, H_,
                            wuc + (size_t)(L * 3 + ls) * 512 * 512, 512, 512, m0, n0, lane);
                    store_tile(uc_pre + (size_t)ls * BH, 512, m0,      n0,      c00, lane);
                    store_tile(uc_pre + (size_t)ls * BH, 512, m0,      n0 + 16, c01, lane);
                    store_tile(uc_pre + (size_t)ls * BH, 512, m0 + 16, n0,      c10, lane);
                    store_tile(uc_pre + (size_t)ls * BH, 512, m0 + 16, n0 + 16, c11, lane);
                } else {
                    // wc_pre [64 x 512], 2x2 blocking: 32 jobs x 64 WMMA
                    const int idx = job - 384;
                    const int mp = idx & 1, np = idx >> 1;     // np in 0..15
                    const int m0 = mp * 32, n0 = np * 32;
                    v8f c00 = {}, c01 = {}, c10 = {}, c11 = {};
                    gemm2x2(c00, c01, c10, c11, inpA, lda,
                            wwc + (size_t)L * 512 * 512, 512, 512, m0, n0, lane);
                    store_tile(wc_pre, 512, m0,      n0,      c00, lane);
                    store_tile(wc_pre, 512, m0,      n0 + 16, c01, lane);
                    store_tile(wc_pre, 512, m0 + 16, n0,      c10, lane);
                    store_tile(wc_pre, 512, m0 + 16, n0 + 16, c11, lane);
                }
            }
            grid_barrier(bar);

            // ---- pointwise phase: gates -> c_new, h_new ----
            for (int idx = gtid; idx < BH; idx += gthreads) {
                const int b = idx >> 9;
                const int j = idx & 511;
                const float* pg = pre_gates + (size_t)b * 1536;
                const float ig = sigmoidf_(pg[j]        + bi2h[L*1536 + j]        + bh2h[L*1536 + j]);
                const float fg = sigmoidf_(pg[512 + j]  + bi2h[L*1536 + 512 + j]  + bh2h[L*1536 + 512 + j]);
                const float og = sigmoidf_(pg[1024 + j] + bi2h[L*1536 + 1024 + j] + bh2h[L*1536 + 1024 + j]);
                float s = 0.f;
#pragma unroll
                for (int l = 0; l < NL_; ++l) {
                    const float g = sigmoidf_(gg_pre[(size_t)l * BH + idx]
                                              + bwg[(L*3 + l)*512 + j] + bug[(L*3 + l)*512 + j]);
                    const float u = uc_pre[(size_t)l * BH + idx] + buc[(L*3 + l)*512 + j];
                    s += g * u;
                }
                const float cand = tanhf(wc_pre[idx] + bwc[L*512 + j] + s);
                const float cn = fg * c_state[(size_t)L * BH + idx] + ig * cand;
                const float hn = og * tanhf(cn);
                c_state[(size_t)L * BH + idx] = cn;
                h_state[(size_t)L * BH + idx] = hn;
                hnext[(size_t)L * BH + idx]   = f2bf(hn);
                if (L == NL_ - 1) out[(size_t)t * BH + idx] = hn;   // out[t][b][j]
            }
            grid_barrier(bar);
        }
    }

    // ---- tail: hT, cT ----
    for (int i = gtid; i < NL_ * BH; i += gthreads) {
        out[(size_t)T_ * BH + i]            = h_state[i];
        out[(size_t)T_ * BH + NL_ * BH + i] = c_state[i];
    }
}

extern "C" void kernel_launch(void* const* d_in, const int* in_sizes, int n_in,
                              void* d_out, int out_size, void* d_ws, size_t ws_size,
                              hipStream_t stream) {
    (void)in_sizes; (void)n_in; (void)out_size; (void)ws_size;
    const float* x    = (const float*)d_in[0];
    const float* h0   = (const float*)d_in[1];
    const float* c0   = (const float*)d_in[2];
    const float* Wi2h = (const float*)d_in[3];
    const float* bi2h = (const float*)d_in[4];
    const float* Wh2h = (const float*)d_in[5];
    const float* bh2h = (const float*)d_in[6];
    const float* Wwc  = (const float*)d_in[7];
    const float* bwc  = (const float*)d_in[8];
    const float* Wwg  = (const float*)d_in[9];
    const float* bwg  = (const float*)d_in[10];
    const float* Wug  = (const float*)d_in[11];
    const float* bug  = (const float*)d_in[12];
    const float* Wuc  = (const float*)d_in[13];
    const float* buc  = (const float*)d_in[14];
    // d_in[15] = length (== T_), compile-time constant here

    hipMemsetAsync(d_ws, 0, 256, stream);   // zero barrier counter (capture-safe)

    GatedFeedbackLSTM_persistent<<<dim3(NWG), dim3(NTHR), 0, stream>>>(
        x, h0, c0, Wi2h, bi2h, Wh2h, bh2h, Wwc, bwc,
        Wwg, bwg, Wug, bug, Wuc, buc,
        (float*)d_out, (char*)d_ws);
}
